// TrulyNeuralGIC_32512902431278
// MI455X (gfx1250) — matrix operand order repaired and checked by reading, not compiled, not run on hardware
//
#include <hip/hip_runtime.h>
#include <hip/hip_bf16.h>
#include <math.h>

typedef __attribute__((ext_vector_type(16))) _Float16 v16h;
typedef __attribute__((ext_vector_type(8)))  float    v8f;

#define MAX_IRQS   256
#define KEY_DIM    64
#define WAVES      4
#define ROWS_PER_WAVE 16
#define ROWS_PER_WG   (WAVES * ROWS_PER_WAVE)

// LDS strides (in halves), padded for bank-conflict-free fragment access
#define HQ_STRIDE  72     // 16 rows x 64 cols (h, then q)
#define P_STRIDE   264    // 16 rows x 256 cols (sim -> exp -> P)
#define VT_STRIDE  264    // 32 rows x 256 cols (V transposed)

static __device__ __forceinline__ v8f wmma_f16(v16h a, v16h b, v8f c) {
  // v_wmma_f32_16x16x32_f16: (neg_a, A, neg_b, B, c_mod, C, reuse_a, reuse_b)
  return __builtin_amdgcn_wmma_f32_16x16x32_f16(false, a, false, b, (short)0, c,
                                                false, false);
}

// A-fragment (16x32 f16, MxK) from an LDS row (row = base of this lane's M row).
// Contiguous K order used by this file: lane m = lane%16; VGPR v holds K pair base = 2v + 16*(lane/16).
static __device__ __forceinline__ v16h load_a_lds(const _Float16* row, int kc, int hh) {
  union { v16h v; unsigned u[8]; } r;
  const _Float16* p = row + kc * 32;
#pragma unroll
  for (int v = 0; v < 8; ++v) {
    int kb = 2 * v + 16 * hh;
    r.u[v] = *(const unsigned*)(p + kb);   // packed pair (K=kb, K=kb+1)
  }
  return r.v;
}

// B-fragment (32x16 f16, KxN) from an LDS row holding V^T[n][k] (k contiguous).
// K order used here (matches load_a_lds): lane n = lane%16; K = 16*(lane/16) + e, e = 0..15 contiguous.
static __device__ __forceinline__ v16h load_b_lds(const _Float16* row, int kc, int hh) {
  union { v16h v; uint4 u[2]; } r;
  const _Float16* p = row + kc * 32 + 16 * hh;
  r.u[0] = *(const uint4*)(p);
  r.u[1] = *(const uint4*)(p + 8);
  return r.v;
}

// B-fragment from 16 consecutive global f16 (pre-converted tables in d_ws)
static __device__ __forceinline__ v16h load_b_gh16(const _Float16* p) {
  union { v16h v; uint4 u[2]; } r;
  r.u[0] = *(const uint4*)(p);
  r.u[1] = *(const uint4*)(p + 8);
  return r.v;
}

// B-fragment from 16 consecutive global f32 (fallback: cvt on the fly)
static __device__ __forceinline__ v16h load_b_gf32(const float* p) {
  v16h b;
#pragma unroll
  for (int i = 0; i < 16; i += 4) {
    float4 f = *(const float4*)(p + i);
    b[i + 0] = (_Float16)f.x; b[i + 1] = (_Float16)f.y;
    b[i + 2] = (_Float16)f.z; b[i + 3] = (_Float16)f.w;
  }
  return b;
}

// B-fragment from global W2 (row-major [64 K][64 N]) -> strided f32 gather + cvt (fallback)
static __device__ __forceinline__ v16h load_b_w2(const float* W2, int t, int kc,
                                                 int hh, int n) {
  v16h b;
  const float* p = W2 + (size_t)(kc * 32 + 16 * hh) * 64 + t * 16 + n;
#pragma unroll
  for (int e = 0; e < 16; ++e) b[e] = (_Float16)p[e * 64];
  return b;
}

// One-shot table prep: keys -> f16 (row-major), W2 -> transposed f16 (w2t[n][k])
__global__ void gic_prep(const float* __restrict__ irq_keys, const float* __restrict__ W2,
                         _Float16* __restrict__ keys16, _Float16* __restrict__ w2t16) {
  const int stride = gridDim.x * blockDim.x;
  const int i0 = blockIdx.x * blockDim.x + threadIdx.x;
  for (int i = i0; i < MAX_IRQS * KEY_DIM; i += stride)
    keys16[i] = (_Float16)irq_keys[i];
  for (int i = i0; i < KEY_DIM * KEY_DIM; i += stride) {
    const int n = i >> 6, k = i & 63;
    w2t16[i] = (_Float16)W2[k * 64 + n];   // w2t[n][k] = W2[k][n]
  }
}

template <bool PRE>
__global__ __launch_bounds__(128)
void gic_kernel(const int* __restrict__ irq_num, const float* __restrict__ cpu_state,
                const float* __restrict__ irq_routing, const float* __restrict__ irq_prior,
                const float* __restrict__ mask_state, const float* __restrict__ irq_keys,
                const float* __restrict__ W1, const float* __restrict__ b1,
                const float* __restrict__ W2, const float* __restrict__ b2,
                const float* __restrict__ temperature,
                const _Float16* __restrict__ keys16, const _Float16* __restrict__ w2t16,
                float* __restrict__ out, int B)
{
  __shared__ float s_w1t[8 * 64];                 // W1 tail rows 256..263
  __shared__ float s_b1[64];
  __shared__ float s_b2[64];
  __shared__ float s_S[WAVES][16];                // softmax denominators
  __shared__ __align__(16) _Float16 s_vt[32 * VT_STRIDE];             // V^T (f16)
  __shared__ __align__(16) _Float16 s_hq[WAVES][16 * HQ_STRIDE];      // h then q
  __shared__ __align__(16) _Float16 s_P [WAVES][16 * P_STRIDE];       // sim -> P

  const int tid  = threadIdx.x;
  const int wave = tid >> 5;
  const int lane = tid & 31;
  const int n16  = lane & 15;    // N-column within a tile (or M row in stage 1/4)
  const int hh   = lane >> 4;    // half-wave index
  const int rowbase = blockIdx.x * ROWS_PER_WG + wave * ROWS_PER_WAVE;

  // ---------------- cooperative init ----------------
  for (int i = tid; i < 512; i += blockDim.x) s_w1t[i] = W1[256 * 64 + i];
  for (int i = tid; i < 64; i += blockDim.x) { s_b1[i] = b1[i]; s_b2[i] = b2[i]; }
  for (int i = tid; i < 32 * 256; i += blockDim.x) {
    int n = i >> 8, k = i & 255;
    float v;
    if      (n < 8)   v = irq_routing[k * 8 + n];
    else if (n < 16)  v = irq_prior  [k * 8 + (n - 8)];
    else if (n == 16) v = mask_state[k];
    else if (n == 17) v = 1.0f;     // ones column -> softmax denominator via GEMM
    else              v = 0.0f;
    s_vt[n * VT_STRIDE + k] = (_Float16)v;
  }
  if (PRE) {                        // warm f16 key table into cache
    for (int i = tid; i < MAX_IRQS / 2; i += blockDim.x)
      __builtin_prefetch(keys16 + (size_t)i * 2 * KEY_DIM, 0, 3);
  } else {
    for (int i = tid; i < MAX_IRQS; i += blockDim.x)
      __builtin_prefetch(irq_keys + (size_t)i * KEY_DIM, 0, 3);
  }
  __syncthreads();

  const float invT = 1.0f / fmaxf(fabsf(temperature[0]), 0.1f);

  // ---------------- stage 1: h = gelu(W1[irq] + cpu @ W1tail + b1) ----------------
  {
    const int m  = n16;
    const int g  = rowbase + m;
    const int gg = (g < B) ? g : (B - 1);
    const int irq = irq_num[gg];
    const bool irq_ok = (irq >= 0) && (irq < MAX_IRQS);
    const float* cp = cpu_state + (size_t)gg * 8;
    float c[8];
#pragma unroll
    for (int j = 0; j < 8; ++j) c[j] = cp[j];
    const float* w1row = W1 + (size_t)(irq_ok ? irq : 0) * 64;
    _Float16* hrow = &s_hq[wave][m * HQ_STRIDE];
    const int k0 = 32 * hh;
#pragma unroll 8
    for (int k = 0; k < 32; ++k) {
      const int kk = k0 + k;
      float x = s_b1[kk] + (irq_ok ? w1row[kk] : 0.0f);
#pragma unroll
      for (int j = 0; j < 8; ++j) x = fmaf(c[j], s_w1t[j * 64 + kk], x);
      const float gel = 0.5f * x * (1.0f + erff(x * 0.70710678118654752f)); // exact GELU
      hrow[kk] = (_Float16)gel;
    }
  }
  __syncthreads();

  // ---------------- stage 2: q = h @ W2 + b2 (WMMA), store q back into s_hq ----------------
  {
    const _Float16* hrow = &s_hq[wave][n16 * HQ_STRIDE];
    const v16h a0 = load_a_lds(hrow, 0, hh);
    const v16h a1 = load_a_lds(hrow, 1, hh);
#pragma unroll
    for (int t = 0; t < 4; ++t) {
      const float bb = s_b2[t * 16 + n16];
      v8f acc = { bb, bb, bb, bb, bb, bb, bb, bb };
      v16h bw0, bw1;
      if (PRE) {
        const _Float16* wrow = w2t16 + (size_t)(t * 16 + n16) * 64;
        bw0 = load_b_gh16(wrow + 16 * hh);
        bw1 = load_b_gh16(wrow + 32 + 16 * hh);
      } else {
        bw0 = load_b_w2(W2, t, 0, hh, n16);
        bw1 = load_b_w2(W2, t, 1, hh, n16);
      }
      acc = wmma_f16(a0, bw0, acc);
      acc = wmma_f16(a1, bw1, acc);
#pragma unroll
      for (int r = 0; r < 8; ++r)   // D layout: M = r + 8*hh, N = t*16 + n16
        s_hq[wave][(r + 8 * hh) * HQ_STRIDE + t * 16 + n16] = (_Float16)acc[r];
    }
  }
  __syncthreads();

  // ---------------- stage 3: sim = q @ keys^T (WMMA), scaled logits -> s_P (f16) ----------------
  {
    const _Float16* qrow = &s_hq[wave][n16 * HQ_STRIDE];
    const v16h qa0 = load_a_lds(qrow, 0, hh);
    const v16h qa1 = load_a_lds(qrow, 1, hh);
    for (int n0 = 0; n0 < 16; ++n0) {
      v16h kb0, kb1;
      if (PRE) {
        const _Float16* krow = keys16 + (size_t)(n0 * 16 + n16) * KEY_DIM;
        kb0 = load_b_gh16(krow + 16 * hh);        // K chunk 0
        kb1 = load_b_gh16(krow + 32 + 16 * hh);   // K chunk 1
      } else {
        const float* krow = irq_keys + (size_t)(n0 * 16 + n16) * KEY_DIM;
        kb0 = load_b_gf32(krow + 16 * hh);
        kb1 = load_b_gf32(krow + 32 + 16 * hh);
      }
      v8f acc = { 0.f, 0.f, 0.f, 0.f, 0.f, 0.f, 0.f, 0.f };
      acc = wmma_f16(qa0, kb0, acc);
      acc = wmma_f16(qa1, kb1, acc);
#pragma unroll
      for (int r = 0; r < 8; ++r)
        s_P[wave][(r + 8 * hh) * P_STRIDE + n0 * 16 + n16] = (_Float16)(acc[r] * invT);
    }
  }
  __syncthreads();

  // ---------------- stage 4: row max + exp in place (2 lanes per row) ----------------
  {
    _Float16* prow = &s_P[wave][n16 * P_STRIDE + 128 * hh];
    float lmax = -3.0e38f;
#pragma unroll
    for (int cidx = 0; cidx < 8; ++cidx) {
      union { v16h v; uint4 u[2]; } x;
      x.u[0] = *(const uint4*)(prow + cidx * 16);
      x.u[1] = *(const uint4*)(prow + cidx * 16 + 8);
#pragma unroll
      for (int e = 0; e < 16; ++e) lmax = fmaxf(lmax, (float)x.v[e]);
    }
    const float rmax = fmaxf(lmax, __shfl_xor(lmax, 16, 32));
#pragma unroll
    for (int cidx = 0; cidx < 8; ++cidx) {
      union { v16h v; uint4 u[2]; } x;
      x.u[0] = *(const uint4*)(prow + cidx * 16);
      x.u[1] = *(const uint4*)(prow + cidx * 16 + 8);
#pragma unroll
      for (int e = 0; e < 16; ++e)
        x.v[e] = (_Float16)__expf((float)x.v[e] - rmax);
      *(uint4*)(prow + cidx * 16)     = x.u[0];
      *(uint4*)(prow + cidx * 16 + 8) = x.u[1];
    }
  }
  __syncthreads();

  // ---------------- stage 5: O[16,32] = P[16,256] @ V (WMMA); normalize; store ----------------
  {
    v8f acc0 = { 0.f, 0.f, 0.f, 0.f, 0.f, 0.f, 0.f, 0.f };
    v8f acc1 = { 0.f, 0.f, 0.f, 0.f, 0.f, 0.f, 0.f, 0.f };
    const _Float16* prow  = &s_P[wave][n16 * P_STRIDE];
    const _Float16* v0row = &s_vt[(n16)      * VT_STRIDE];   // N tile 0: cols 0..15
    const _Float16* v1row = &s_vt[(16 + n16) * VT_STRIDE];   // N tile 1: cols 16..31
#pragma unroll
    for (int kc = 0; kc < 8; ++kc) {
      const v16h pa  = load_a_lds(prow,  kc, hh);
      const v16h vb0 = load_b_lds(v0row, kc, hh);
      const v16h vb1 = load_b_lds(v1row, kc, hh);
      acc0 = wmma_f16(pa, vb0, acc0);
      acc1 = wmma_f16(pa, vb1, acc1);
    }
    // column 17 (ones) = softmax denominator S, held by lanes with n16 == 1
    if (n16 == 1) {
#pragma unroll
      for (int r = 0; r < 8; ++r) s_S[wave][r + 8 * hh] = acc1[r];
    }
    __syncthreads();

    const size_t B8 = (size_t)B * 8, B16 = (size_t)B * 16;
#pragma unroll
    for (int r = 0; r < 8; ++r) {
      const int m = r + 8 * hh;
      const int g = rowbase + m;
      if (g < B) {
        const float inv = 1.0f / s_S[wave][m];
        if (n16 < 8) out[(size_t)g * 8 + n16]            = acc0[r] * inv; // handler
        else         out[B8 + (size_t)g * 8 + (n16 - 8)] = acc0[r] * inv; // priority
        if (n16 == 0) {                                   // col 16 = mask_value
          const float mv = acc1[r] * inv;
          out[B16 + g] = 1.0f / (1.0f + __expf(mv));      // sigmoid(-mv)
        }
      }
    }
  }
}

extern "C" void kernel_launch(void* const* d_in, const int* in_sizes, int n_in,
                              void* d_out, int out_size, void* d_ws, size_t ws_size,
                              hipStream_t stream) {
  const int*   irq_num   = (const int*)  d_in[0];
  const float* cpu_state = (const float*)d_in[1];
  const float* routing   = (const float*)d_in[2];
  const float* prior     = (const float*)d_in[3];
  const float* maskst    = (const float*)d_in[4];
  const float* keys      = (const float*)d_in[5];
  const float* W1        = (const float*)d_in[6];
  const float* b1        = (const float*)d_in[7];
  const float* W2        = (const float*)d_in[8];
  const float* b2        = (const float*)d_in[9];
  const float* temp      = (const float*)d_in[10];
  float* out = (float*)d_out;
  const int B = in_sizes[0];
  const int grid = (B + ROWS_PER_WG - 1) / ROWS_PER_WG;

  const size_t need = (size_t)(MAX_IRQS * KEY_DIM + KEY_DIM * KEY_DIM) * sizeof(_Float16);
  if (ws_size >= need) {
    _Float16* keys16 = (_Float16*)d_ws;
    _Float16* w2t16  = keys16 + MAX_IRQS * KEY_DIM;
    gic_prep<<<dim3(40), dim3(256), 0, stream>>>(keys, W2, keys16, w2t16);
    gic_kernel<true><<<dim3(grid), dim3(128), 0, stream>>>(
        irq_num, cpu_state, routing, prior, maskst, keys, W1, b1, W2, b2, temp,
        keys16, w2t16, out, B);
  } else {
    gic_kernel<false><<<dim3(grid), dim3(128), 0, stream>>>(
        irq_num, cpu_state, routing, prior, maskst, keys, W1, b1, W2, b2, temp,
        (const _Float16*)nullptr, (const _Float16*)nullptr, out, B);
  }
  (void)n_in; (void)out_size;
}
